// MultiModalFusionModelWithAblation_28535762714793
// MI455X (gfx1250) — matrix operand order repaired and checked by reading, not compiled, not run on hardware
//
#include <hip/hip_runtime.h>
#include <hip/hip_bf16.h>

#define BBATCH 16384
#define HDIM   512
#define NMODS  5

typedef __attribute__((ext_vector_type(16))) __bf16 v16bf;
typedef __attribute__((ext_vector_type(8)))  float  v8f;
typedef __attribute__((ext_vector_type(4)))  unsigned int u32x4;
typedef __attribute__((ext_vector_type(8)))  int i32x8;
typedef __attribute__((ext_vector_type(4)))  int i32x4;

#if __has_builtin(__builtin_amdgcn_tensor_load_to_lds)
#define HAVE_TDM 1
#else
#define HAVE_TDM 0
#endif
#if __has_builtin(__builtin_amdgcn_s_wait_tensorcnt)
#define HAVE_TENSORCNT 1
#else
#define HAVE_TENSORCNT 0
#endif

// ---------------- helpers ----------------
static __device__ __forceinline__ unsigned short f2bf(float f) {
  unsigned int u = __float_as_uint(f);
  unsigned int r = u + 0x7FFFu + ((u >> 16) & 1u);
  return (unsigned short)(r >> 16);
}
static __device__ __forceinline__ unsigned int packbf2(float a, float b) {
#if __has_builtin(__builtin_amdgcn_cvt_pk_bf16_f32)
  auto r = __builtin_amdgcn_cvt_pk_bf16_f32(a, b);
  return __builtin_bit_cast(unsigned int, r);
#else
  return (unsigned int)f2bf(a) | ((unsigned int)f2bf(b) << 16);
#endif
}
static __device__ __forceinline__ float bf2f(unsigned short h) {
  return __uint_as_float(((unsigned int)h) << 16);
}
static __device__ __forceinline__ float wred32(float v) {
#pragma unroll
  for (int s = 16; s >= 1; s >>= 1) v += __shfl_xor(v, s, 32);
  return v;
}
static __device__ __forceinline__ void unpack8(uint4 p, float* f) {
  unsigned int w0 = p.x, w1 = p.y, w2 = p.z, w3 = p.w;
  f[0] = bf2f((unsigned short)(w0 & 0xffff)); f[1] = bf2f((unsigned short)(w0 >> 16));
  f[2] = bf2f((unsigned short)(w1 & 0xffff)); f[3] = bf2f((unsigned short)(w1 >> 16));
  f[4] = bf2f((unsigned short)(w2 & 0xffff)); f[5] = bf2f((unsigned short)(w2 >> 16));
  f[6] = bf2f((unsigned short)(w3 & 0xffff)); f[7] = bf2f((unsigned short)(w3 >> 16));
}
static __device__ __forceinline__ uint4 pack8(const float* f) {
  uint4 p;
  p.x = packbf2(f[0], f[1]);
  p.y = packbf2(f[2], f[3]);
  p.z = packbf2(f[4], f[5]);
  p.w = packbf2(f[6], f[7]);
  return p;
}

// ---------------- weight convert + transpose: dst[n*K+k] = bf16(src[k*ld + col_off + n]) ----------------
__global__ __launch_bounds__(256) void cvt_tr(const float* __restrict__ src, int src_ld, int col_off,
                                              unsigned short* __restrict__ dst, int K, int N) {
  long i = (long)blockIdx.x * 256 + threadIdx.x;
  long total = (long)K * N;
  if (i >= total) return;
  int n = (int)(i / K);
  int k = (int)(i % K);
  dst[i] = f2bf(src[(size_t)k * src_ld + col_off + n]);
}

// ---------------- generic bf16 WMMA GEMM: C[M,N] = act(A[M,K] @ Bt^T + bias) ----------------
// A: fp32 or bf16, row stride lda (elements). Bt: bf16 [N][K]. Tiles: 32x256, K-step 32.
// B tile staged either by Tensor Data Mover (wave 0 issues TDM, s_wait_tensorcnt) or vector loads.
__global__ __launch_bounds__(256) void gemm_bf16_wmma(
    const void* __restrict__ Aptr, int a_is_f32, long lda,
    const unsigned short* __restrict__ Bt,
    const float* __restrict__ bias, int relu,
    unsigned short* __restrict__ C, long ldc,
    float* __restrict__ C32, long ldc32,
    int M, int N, int K, int use_tdm) {
  (void)M; (void)N;
  __shared__ unsigned short As[32][48];
  __shared__ unsigned short Bs[256][48];   // 96B row pitch == TDM pad: 16 dwords data + 8 dwords pad
  const int tid  = threadIdx.x;
  const int w    = tid >> 5;
  const int lane = tid & 31;
  const int m16  = lane & 15;
  const int hi   = lane >> 4;
  const int row0 = blockIdx.y * 32;
  const int n0   = blockIdx.x * 256;

  v8f acc[2][2] = {};

  const int aM = tid >> 3;          // 0..31
  const int aC = (tid & 7) << 2;    // 0,4,...,28

#if HAVE_TDM
  // D# group1 invariant part: mask=0, data_size=2B, pad_enable, pad_interval=16dw, pad_amount=8dw
  const unsigned g1dw0 = (1u << 16) | (1u << 20) | (3u << 22) | (7u << 25);
  const unsigned td0 = (unsigned)K;      // tensor_dim0 (elements)
  const unsigned td1 = 256u;             // tensor_dim1 (rows of this tile view)
  const unsigned g1dw1 = (td0 & 0xffffu) << 16;
  const unsigned g1dw2 = (td0 >> 16) | ((td1 & 0xffffu) << 16);
  const unsigned g1dw3 = (td1 >> 16) | (32u << 16);  // tile_dim0 = 32 elements (64B)
  const unsigned g1dw4 = 256u;                       // tile_dim1 = 256, tile_dim2 = 0
  const unsigned g1dw5 = (unsigned)K;                // tensor_dim0_stride low32 (elements)
  const unsigned ldsB = (unsigned)(unsigned long long)(const void*)&Bs[0][0];
#endif

  for (int kk = 0; kk < K; kk += 32) {
    __syncthreads();
    // ---- stage A tile (32 x 32) ----
    if (a_is_f32) {
      const float* A = (const float*)Aptr;
      float4 f = *(const float4*)(A + (size_t)(row0 + aM) * lda + kk + aC);
      uint2 packed;
      packed.x = packbf2(f.x, f.y);
      packed.y = packbf2(f.z, f.w);
      *(uint2*)&As[aM][aC] = packed;
    } else {
      const unsigned short* A = (const unsigned short*)Aptr;
      *(uint2*)&As[aM][aC] = *(const uint2*)(A + (size_t)(row0 + aM) * lda + kk + aC);
    }
    // ---- stage B tile (256 rows of Bt, 32 k each) ----
#if HAVE_TDM
    if (use_tdm) {
      if (w == 0) {  // wave-uniform: only wave 0 drives the Tensor Data Mover
        unsigned long long ga =
            (unsigned long long)(const void*)(Bt + (size_t)n0 * K + kk);
        u32x4 g0;
        g0.x = 1u;                                   // count=1, user descriptor
        g0.y = ldsB;                                 // lds_addr
        g0.z = (unsigned)ga;                         // global_addr[31:0]
        g0.w = (unsigned)((ga >> 32) & 0x1FFFFFFu) | (2u << 30);  // addr[56:32] | type=2
        i32x8 g1;
        g1.s0 = (int)g1dw0; g1.s1 = (int)g1dw1; g1.s2 = (int)g1dw2; g1.s3 = (int)g1dw3;
        g1.s4 = (int)g1dw4; g1.s5 = (int)g1dw5; g1.s6 = 0; g1.s7 = 0;
        i32x4 z4 = (i32x4)0;
#if __clang_major__ >= 23
        i32x8 z8 = (i32x8)0;
        __builtin_amdgcn_tensor_load_to_lds(g0, g1, z4, z4, z8, 0);
#else
        __builtin_amdgcn_tensor_load_to_lds(g0, g1, z4, z4, 0);
#endif
#if HAVE_TENSORCNT
        __builtin_amdgcn_s_wait_tensorcnt(0);
#endif
      }
    } else
#endif
    {
      const unsigned short* brow = Bt + (size_t)(n0 + tid) * K + kk;
      *(uint4*)&Bs[tid][0]  = *(const uint4*)(brow);
      *(uint4*)&Bs[tid][8]  = *(const uint4*)(brow + 8);
      *(uint4*)&Bs[tid][16] = *(const uint4*)(brow + 16);
      *(uint4*)&Bs[tid][24] = *(const uint4*)(brow + 24);
      if (kk + 32 < K) __builtin_prefetch(brow + 32, 0, 1);
    }
    __syncthreads();

    // ---- fragments ----
    const int kh = hi << 3;   // A: chunks at k=kh and k=16+kh
    const int k0 = hi << 3;   // B: chunks at k=k0 and k=16+k0
    union { uint4 q[2]; v16bf v; } af[2], bfr[2];
#pragma unroll
    for (int r = 0; r < 2; ++r) {
      af[r].q[0] = *(const uint4*)&As[r * 16 + m16][kh];
      af[r].q[1] = *(const uint4*)&As[r * 16 + m16][16 + kh];
    }
#pragma unroll
    for (int c = 0; c < 2; ++c) {
      const int n = (w << 5) + (c << 4) + m16;
      bfr[c].q[0] = *(const uint4*)&Bs[n][k0];
      bfr[c].q[1] = *(const uint4*)&Bs[n][16 + k0];
    }
#pragma unroll
    for (int r = 0; r < 2; ++r)
#pragma unroll
      for (int c = 0; c < 2; ++c)
        acc[r][c] = __builtin_amdgcn_wmma_f32_16x16x32_bf16(
            false, af[r].v, false, bfr[c].v, (short)0, acc[r][c], false, false);
  }

  // ---- epilogue ----
#pragma unroll
  for (int r = 0; r < 2; ++r) {
#pragma unroll
    for (int c = 0; c < 2; ++c) {
      const int col = n0 + (w << 5) + (c << 4) + m16;
      const float bv = bias ? bias[col] : 0.0f;
#pragma unroll
      for (int v = 0; v < 8; ++v) {
        const int row = row0 + r * 16 + v + (hi << 3);
        float x = acc[r][c][v] + bv;
        if (relu) x = fmaxf(x, 0.0f);
        if (C)   C[(size_t)row * ldc + col] = f2bf(x);
        if (C32) C32[(size_t)row * ldc32 + col] = x;
      }
    }
  }
}

// ---------------- LayerNorm over H=512, optional residual; params indexed by row%5 ----------------
__global__ __launch_bounds__(256) void ln_rows(
    const unsigned short* __restrict__ in, const unsigned short* __restrict__ res,
    const float* __restrict__ gg, const float* __restrict__ bb,
    unsigned short* __restrict__ out, int nrows) {
  int row  = blockIdx.x * 8 + (threadIdx.x >> 5);
  int lane = threadIdx.x & 31;
  if (row >= nrows) return;
  int mod = row % NMODS;
  const unsigned short* ip = in + (size_t)row * HDIM + lane * 16;
  float v[16];
  unpack8(*(const uint4*)ip, v);
  unpack8(*(const uint4*)(ip + 8), v + 8);
  if (res) {
    const unsigned short* rp = res + (size_t)row * HDIM + lane * 16;
    float rv[16];
    unpack8(*(const uint4*)rp, rv);
    unpack8(*(const uint4*)(rp + 8), rv + 8);
#pragma unroll
    for (int j = 0; j < 16; ++j) v[j] += rv[j];
  }
  float s = 0.f, s2 = 0.f;
#pragma unroll
  for (int j = 0; j < 16; ++j) { s += v[j]; s2 += v[j] * v[j]; }
  s = wred32(s); s2 = wred32(s2);
  float mean = s * (1.0f / 512.0f);
  float var  = s2 * (1.0f / 512.0f) - mean * mean;
  float rstd = rsqrtf(var + 1e-5f);
  const float* gptr = gg + (size_t)mod * HDIM + lane * 16;
  const float* bptr = bb + (size_t)mod * HDIM + lane * 16;
  float o[16];
#pragma unroll
  for (int j = 0; j < 16; ++j) o[j] = (v[j] - mean) * rstd * gptr[j] + bptr[j];
  unsigned short* op = out + (size_t)row * HDIM + lane * 16;
  *(uint4*)op = pack8(o);
  *(uint4*)(op + 8) = pack8(o + 8);
}

// ---------------- GAT: s1/s2 dots, leaky-softmax over 5x5, ctx = attn @ Wh ----------------
__global__ __launch_bounds__(256) void gat_ctx(const unsigned short* __restrict__ Wh,
                                               const float* __restrict__ gat_a,
                                               unsigned short* __restrict__ ctx) {
  int t   = blockIdx.x * 256 + threadIdx.x;
  int s   = t >> 2;          // sample
  int seg = t & 3;           // 128-col segment
  const float* a1 = gat_a;
  const float* a2 = gat_a + HDIM;
  const unsigned short* base = Wh + (size_t)s * (NMODS * HDIM);
  float s1[NMODS], s2[NMODS];
#pragma unroll
  for (int j = 0; j < NMODS; ++j) { s1[j] = 0.f; s2[j] = 0.f; }
  const int c0 = seg * 128;
#pragma unroll
  for (int j = 0; j < NMODS; ++j) {
    const unsigned short* rowp = base + (size_t)j * HDIM + c0;
    for (int cc = 0; cc < 128; cc += 8) {
      float f[8];
      unpack8(*(const uint4*)(rowp + cc), f);
#pragma unroll
      for (int u = 0; u < 8; ++u) {
        s1[j] += f[u] * a1[c0 + cc + u];
        s2[j] += f[u] * a2[c0 + cc + u];
      }
    }
  }
#pragma unroll
  for (int j = 0; j < NMODS; ++j) {
    s1[j] += __shfl_xor(s1[j], 1, 32); s1[j] += __shfl_xor(s1[j], 2, 32);
    s2[j] += __shfl_xor(s2[j], 1, 32); s2[j] += __shfl_xor(s2[j], 2, 32);
  }
  float attn[NMODS][NMODS];
#pragma unroll
  for (int i = 0; i < NMODS; ++i) {
    float m = -1e30f;
#pragma unroll
    for (int j = 0; j < NMODS; ++j) {
      float e = s1[i] + s2[j];
      e = (e > 0.f) ? e : 0.2f * e;
      attn[i][j] = e;
      m = fmaxf(m, e);
    }
    float sum = 0.f;
#pragma unroll
    for (int j = 0; j < NMODS; ++j) { attn[i][j] = expf(attn[i][j] - m); sum += attn[i][j]; }
    float inv = 1.0f / sum;
#pragma unroll
    for (int j = 0; j < NMODS; ++j) attn[i][j] *= inv;
  }
  unsigned short* outb = ctx + (size_t)s * (NMODS * HDIM);
  for (int cc = 0; cc < 128; cc += 8) {
    float wv[NMODS][8];
#pragma unroll
    for (int j = 0; j < NMODS; ++j)
      unpack8(*(const uint4*)(base + (size_t)j * HDIM + c0 + cc), wv[j]);
#pragma unroll
    for (int i = 0; i < NMODS; ++i) {
      float o[8];
#pragma unroll
      for (int u = 0; u < 8; ++u) {
        float acc = 0.f;
#pragma unroll
        for (int j = 0; j < NMODS; ++j) acc += attn[i][j] * wv[j][u];
        o[u] = acc;
      }
      *(uint4*)(outb + (size_t)i * HDIM + c0 + cc) = pack8(o);
    }
  }
}

// ---------------- MHA pooling: per (sample, query, head): softmax over 5 keys ----------------
__global__ __launch_bounds__(256) void pool_attn(const unsigned short* __restrict__ kv,
                                                 const float* __restrict__ qe,
                                                 const float* __restrict__ qp,
                                                 unsigned short* __restrict__ pe,
                                                 unsigned short* __restrict__ pp) {
  int t     = blockIdx.x * 256 + threadIdx.x;
  int head  = t & 7;
  int query = (t >> 3) & 1;
  int s     = t >> 4;
  const float* q = query ? qp : qe;
  const unsigned short* base = kv + (size_t)s * (NMODS * 1024);
  float sc[NMODS];
#pragma unroll
  for (int n = 0; n < NMODS; ++n) {
    float dot = 0.f;
    const unsigned short* kp = base + (size_t)n * 1024 + head * 64;
    for (int d = 0; d < 64; d += 8) {
      float f[8];
      unpack8(*(const uint4*)(kp + d), f);
#pragma unroll
      for (int u = 0; u < 8; ++u) dot += f[u] * q[head * 64 + d + u];
    }
    sc[n] = dot * 0.125f;  // 1/sqrt(64)
  }
  float m = -1e30f;
#pragma unroll
  for (int n = 0; n < NMODS; ++n) m = fmaxf(m, sc[n]);
  float sum = 0.f;
#pragma unroll
  for (int n = 0; n < NMODS; ++n) { sc[n] = expf(sc[n] - m); sum += sc[n]; }
  float inv = 1.0f / sum;
  unsigned short* dst = (query ? pp : pe) + (size_t)s * HDIM + head * 64;
  for (int d = 0; d < 64; d += 8) {
    float o[8];
#pragma unroll
    for (int u = 0; u < 8; ++u) o[u] = 0.f;
#pragma unroll
    for (int n = 0; n < NMODS; ++n) {
      float f[8];
      unpack8(*(const uint4*)(base + (size_t)n * 1024 + 512 + head * 64 + d), f);
#pragma unroll
      for (int u = 0; u < 8; ++u) o[u] += sc[n] * f[u];
    }
#pragma unroll
    for (int u = 0; u < 8; ++u) o[u] *= inv;
    *(uint4*)(dst + d) = pack8(o);
  }
}

// ---------------- tiny: project queries through Wq ----------------
__global__ __launch_bounds__(256) void qproj(const float* __restrict__ eq, const float* __restrict__ pq,
                                             const float* __restrict__ w, const float* __restrict__ b,
                                             float* __restrict__ qe, float* __restrict__ qp) {
  int j = blockIdx.x * 256 + threadIdx.x;
  if (j >= HDIM) return;
  float ae = 0.f, ap = 0.f;
  for (int i = 0; i < HDIM; ++i) {
    float wv = w[(size_t)i * 1536 + j];
    ae += eq[i] * wv;
    ap += pq[i] * wv;
  }
  qe[j] = ae + b[j];
  qp[j] = ap + b[j];
}

// ---------------- tiny: guide vector norms (7 emo + 5 pkl) ----------------
__global__ void guide_norms(const float* __restrict__ ge, const float* __restrict__ gp,
                            float* __restrict__ norms) {
  int row  = threadIdx.x >> 5;
  int lane = threadIdx.x & 31;
  if (row >= 12) return;
  const float* src = (row < 7) ? (ge + (size_t)row * HDIM) : (gp + (size_t)(row - 7) * HDIM);
  float ss = 0.f;
  for (int j = lane; j < HDIM; j += 32) ss += src[j] * src[j];
  ss = wred32(ss);
  if (lane == 0) norms[row] = sqrtf(ss);
}

// ---------------- final heads + cosine guides ----------------
__global__ __launch_bounds__(256) void final_stage(
    const float* __restrict__ re, const float* __restrict__ rp,
    const float* __restrict__ elo, const float* __restrict__ pso,
    const float* __restrict__ elp_w, const float* __restrict__ elp_b,
    const float* __restrict__ plp_w, const float* __restrict__ plp_b,
    const float* __restrict__ ehw, const float* __restrict__ ehb,
    const float* __restrict__ phw, const float* __restrict__ phb,
    const float* __restrict__ ge, const float* __restrict__ gp,
    const float* __restrict__ norms, float* __restrict__ out) {
  int s    = blockIdx.x * 8 + (threadIdx.x >> 5);
  int lane = threadIdx.x & 31;
  float m7v = 0.f, m5v = 0.f;
  if (lane < 7) {
#pragma unroll
    for (int m = 0; m < NMODS; ++m) m7v += elo[(size_t)m * BBATCH * 7 + (size_t)s * 7 + lane];
    m7v *= 0.2f;
  }
  if (lane < 5) {
#pragma unroll
    for (int m = 0; m < NMODS; ++m) m5v += pso[(size_t)m * BBATCH * 5 + (size_t)s * 5 + lane];
    m5v *= 0.2f;
  }
  float m7[7], m5[5];
#pragma unroll
  for (int c = 0; c < 7; ++c) m7[c] = __shfl(m7v, c, 32);
#pragma unroll
  for (int c = 0; c < 5; ++c) m5[c] = __shfl(m5v, c, 32);

  const int d0 = lane * 16;
  float reprE[16], reprP[16];
#pragma unroll
  for (int j = 0; j < 16; ++j) {
    int d = d0 + j;
    float v = re[(size_t)s * HDIM + d] + elp_b[d];
#pragma unroll
    for (int c = 0; c < 7; ++c) v += m7[c] * elp_w[(size_t)c * HDIM + d];
    reprE[j] = v;
    float u = rp[(size_t)s * HDIM + d] + plp_b[d];
#pragma unroll
    for (int c = 0; c < 5; ++c) u += m5[c] * plp_w[(size_t)c * HDIM + d];
    reprP[j] = u;
  }
  float hdE[7], gdE[7], hdP[5], gdP[5], nE = 0.f, nP = 0.f;
#pragma unroll
  for (int c = 0; c < 7; ++c) { hdE[c] = 0.f; gdE[c] = 0.f; }
#pragma unroll
  for (int c = 0; c < 5; ++c) { hdP[c] = 0.f; gdP[c] = 0.f; }
#pragma unroll
  for (int j = 0; j < 16; ++j) {
    int d = d0 + j;
    float x = reprE[j];
    nE += x * x;
#pragma unroll
    for (int c = 0; c < 7; ++c) { hdE[c] += x * ehw[(size_t)d * 7 + c]; gdE[c] += x * ge[(size_t)c * HDIM + d]; }
    float y = reprP[j];
    nP += y * y;
#pragma unroll
    for (int c = 0; c < 5; ++c) { hdP[c] += y * phw[(size_t)d * 5 + c]; gdP[c] += y * gp[(size_t)c * HDIM + d]; }
  }
#pragma unroll
  for (int c = 0; c < 7; ++c) { hdE[c] = wred32(hdE[c]); gdE[c] = wred32(gdE[c]); }
#pragma unroll
  for (int c = 0; c < 5; ++c) { hdP[c] = wred32(hdP[c]); gdP[c] = wred32(gdP[c]); }
  nE = wred32(nE);
  nP = wred32(nP);
  float rnE = fmaxf(sqrtf(nE), 1e-8f);
  float rnP = fmaxf(sqrtf(nP), 1e-8f);
  if (lane < 7) {
    float pred = 0.f, gdot = 0.f;
#pragma unroll
    for (int c = 0; c < 7; ++c) if (lane == c) { pred = hdE[c]; gdot = gdE[c]; }
    pred += ehb[lane];
    float cosv = gdot / (rnE * fmaxf(norms[lane], 1e-8f));
    out[(size_t)s * 12 + lane] = 0.5f * (pred + cosv);
  }
  if (lane < 5) {
    float pred = 0.f, gdot = 0.f;
#pragma unroll
    for (int c = 0; c < 5; ++c) if (lane == c) { pred = hdP[c]; gdot = gdP[c]; }
    pred = 1.0f / (1.0f + expf(-(pred + phb[lane])));
    float cosv = gdot / (rnP * fmaxf(norms[7 + lane], 1e-8f));
    float sc = 1.0f / (1.0f + expf(-cosv));
    out[(size_t)s * 12 + 7 + lane] = 0.5f * (pred + sc);
  }
}

// ---------------- host launcher ----------------
extern "C" void kernel_launch(void* const* d_in, const int* in_sizes, int n_in,
                              void* d_out, int out_size, void* d_ws, size_t ws_size,
                              hipStream_t stream) {
  (void)in_sizes; (void)n_in; (void)out_size;
  const int IN[5] = {2048, 1024, 1536, 512, 512};

  char* ws = (char*)d_ws;
  size_t off = 0;
  auto alloc = [&](size_t bytes) -> char* {
    char* p = ws + off;
    off += (bytes + 255) & ~(size_t)255;
    return p;
  };
  unsigned short* WpT[5];
  for (int i = 0; i < 5; ++i) WpT[i] = (unsigned short*)alloc((size_t)512 * IN[i] * 2);
  unsigned short* aW1T  = (unsigned short*)alloc((size_t)5 * 256 * 512 * 2);
  unsigned short* aW2T  = (unsigned short*)alloc((size_t)5 * 512 * 256 * 2);
  unsigned short* gatWT = (unsigned short*)alloc((size_t)512 * 512 * 2);
  unsigned short* WkvT  = (unsigned short*)alloc((size_t)1024 * 512 * 2);
  unsigned short* outWT = (unsigned short*)alloc((size_t)512 * 512 * 2);
  float* qe    = (float*)alloc(512 * 4);
  float* qp    = (float*)alloc(512 * 4);
  float* norms = (float*)alloc(64);
  unsigned short* buf1  = (unsigned short*)alloc((size_t)BBATCH * 5 * 512 * 2);
  unsigned short* bufH  = (unsigned short*)alloc((size_t)BBATCH * 5 * 1024 * 2);
  unsigned short* tbuf  = (unsigned short*)alloc((size_t)BBATCH * 256 * 2);
  unsigned short* poolE = (unsigned short*)alloc((size_t)BBATCH * 512 * 2);
  unsigned short* poolP = (unsigned short*)alloc((size_t)BBATCH * 512 * 2);
  float* reprE = (float*)alloc((size_t)BBATCH * 512 * 4);
  float* reprP = (float*)alloc((size_t)BBATCH * 512 * 4);
  if (off > ws_size) return;

  const float* feat[5] = {(const float*)d_in[0], (const float*)d_in[2], (const float*)d_in[4],
                          (const float*)d_in[6], (const float*)d_in[8]};
  const float* Wp[5]   = {(const float*)d_in[1], (const float*)d_in[3], (const float*)d_in[5],
                          (const float*)d_in[7], (const float*)d_in[9]};
  const float* bp   = (const float*)d_in[10];
  const float* ln1g = (const float*)d_in[11];
  const float* ln1b = (const float*)d_in[12];
  const float* aW1  = (const float*)d_in[13];
  const float* ab1  = (const float*)d_in[14];
  const float* aW2  = (const float*)d_in[15];
  const float* ab2  = (const float*)d_in[16];
  const float* ln2g = (const float*)d_in[17];
  const float* ln2b = (const float*)d_in[18];
  const float* gatW = (const float*)d_in[19];
  const float* gata = (const float*)d_in[20];
  const float* eqv  = (const float*)d_in[21];
  const float* pqv  = (const float*)d_in[22];
  const float* mhaw = (const float*)d_in[23];
  const float* mhab = (const float*)d_in[24];
  const float* outw = (const float*)d_in[25];
  const float* outb = (const float*)d_in[26];
  const float* ehw  = (const float*)d_in[27];
  const float* ehb  = (const float*)d_in[28];
  const float* phw  = (const float*)d_in[29];
  const float* phb  = (const float*)d_in[30];
  const float* ge   = (const float*)d_in[31];
  const float* gp   = (const float*)d_in[32];
  const float* elpw = (const float*)d_in[33];
  const float* elpb = (const float*)d_in[34];
  const float* plpw = (const float*)d_in[35];
  const float* plpb = (const float*)d_in[36];
  const float* elo  = (const float*)d_in[37];
  const float* pso  = (const float*)d_in[38];

  auto cvt = [&](unsigned short* dst, const float* src, int src_ld, int col_off, int K, int N) {
    long total = (long)K * N;
    int blocks = (int)((total + 255) / 256);
    cvt_tr<<<blocks, 256, 0, stream>>>(src, src_ld, col_off, dst, K, N);
  };
  for (int i = 0; i < 5; ++i) cvt(WpT[i], Wp[i], 512, 0, IN[i], 512);
  for (int i = 0; i < 5; ++i) cvt(aW1T + (size_t)i * 256 * 512, aW1 + (size_t)i * 512 * 256, 256, 0, 512, 256);
  for (int i = 0; i < 5; ++i) cvt(aW2T + (size_t)i * 512 * 256, aW2 + (size_t)i * 256 * 512, 512, 0, 256, 512);
  cvt(gatWT, gatW, 512, 0, 512, 512);
  cvt(WkvT, mhaw, 1536, 512, 512, 1024);
  cvt(outWT, outw, 512, 0, 512, 512);

  qproj<<<2, 256, 0, stream>>>(eqv, pqv, mhaw, mhab, qe, qp);
  guide_norms<<<1, 384, 0, stream>>>(ge, gp, norms);

  const int USE_TDM = 1;

  // stage A: per-modality projection + ReLU (fused) -> buf1 slots; then LN1 -> bufH
  for (int i = 0; i < 5; ++i)
    gemm_bf16_wmma<<<dim3(2, 512), 256, 0, stream>>>(feat[i], 1, (long)IN[i], WpT[i],
        bp + (size_t)i * 512, 1, buf1 + (size_t)i * 512, 2560, nullptr, 0, BBATCH, 512, IN[i], USE_TDM);
  ln_rows<<<BBATCH * 5 / 8, 256, 0, stream>>>(buf1, nullptr, ln1g, ln1b, bufH, BBATCH * 5);

  // adapters: t = relu(h@aW1+b); a = t@aW2+b -> buf1; LN2(h + a) -> buf1 (x)
  for (int i = 0; i < 5; ++i) {
    gemm_bf16_wmma<<<dim3(1, 512), 256, 0, stream>>>(bufH + (size_t)i * 512, 0, 2560,
        aW1T + (size_t)i * 256 * 512, ab1 + (size_t)i * 256, 1, tbuf, 256, nullptr, 0,
        BBATCH, 256, 512, USE_TDM);
    gemm_bf16_wmma<<<dim3(2, 512), 256, 0, stream>>>(tbuf, 0, 256,
        aW2T + (size_t)i * 512 * 256, ab2 + (size_t)i * 512, 0, buf1 + (size_t)i * 512, 2560,
        nullptr, 0, BBATCH, 512, 256, USE_TDM);
  }
  ln_rows<<<BBATCH * 5 / 8, 256, 0, stream>>>(buf1, bufH, ln2g, ln2b, buf1, BBATCH * 5);

  // GAT: Wh = x @ gat_W -> bufH; softmax-context -> buf1 (ctx)
  gemm_bf16_wmma<<<dim3(2, 2560), 256, 0, stream>>>(buf1, 0, 512, gatWT, nullptr, 0,
      bufH, 512, nullptr, 0, BBATCH * 5, 512, 512, USE_TDM);
  gat_ctx<<<BBATCH * 4 / 256, 256, 0, stream>>>(bufH, gata, buf1);

  // MHA: kv = ctx @ [Wk|Wv] + b -> bufH; pooling -> poolE/poolP; out-proj -> reprE/reprP (fp32)
  gemm_bf16_wmma<<<dim3(4, 2560), 256, 0, stream>>>(buf1, 0, 512, WkvT, mhab + 512, 0,
      bufH, 1024, nullptr, 0, BBATCH * 5, 1024, 512, USE_TDM);
  pool_attn<<<BBATCH * 16 / 256, 256, 0, stream>>>(bufH, qe, qp, poolE, poolP);
  gemm_bf16_wmma<<<dim3(2, 512), 256, 0, stream>>>(poolE, 0, 512, outWT, outb, 0,
      nullptr, 0, reprE, 512, BBATCH, 512, 512, USE_TDM);
  gemm_bf16_wmma<<<dim3(2, 512), 256, 0, stream>>>(poolP, 0, 512, outWT, outb, 0,
      nullptr, 0, reprP, 512, BBATCH, 512, 512, USE_TDM);

  // final heads + cosine guidance
  final_stage<<<BBATCH / 8, 256, 0, stream>>>(reprE, reprP, elo, pso, elpw, elpb, plpw, plpb,
      ehw, ehb, phw, phb, ge, gp, norms, (float*)d_out);
}
